// EquivariantCrossAttentionNeF_27908697489438
// MI455X (gfx1250) — matrix operand order repaired and checked by reading, not compiled, not run on hardware
//
#include <hip/hip_runtime.h>
#include <hip/hip_bf16.h>

// ---------------------------------------------------------------------------
// MI455X (gfx1250) implementation.
// Heavy math = per-pair embedding MLPs -> batched GEMMs on v_wmma_f32_16x16x32_f16
// (f16 inputs, f32 accumulate). Weights pre-converted to f16 and pre-swizzled
// into the WMMA B-fragment lane layout so each fragment is one contiguous
// 32-byte load. Attention kernel holds emb weights in LDS (192KB of 320KB/WGP),
// staged with CDNA5 async global->LDS copies when the toolchain exposes them.
// ---------------------------------------------------------------------------

typedef __attribute__((ext_vector_type(16))) _Float16 v16h;
typedef __attribute__((ext_vector_type(8)))  _Float16 v8h;
typedef __attribute__((ext_vector_type(8)))  float    v8f;

#define WMMA_F16(a, b, c) \
  __builtin_amdgcn_wmma_f32_16x16x32_f16(false, (a), false, (b), (short)0, (c), false, false)

// CDNA5 async global->LDS copy (ASYNCcnt-tracked, no VGPR staging).
// Probe result: param 0 is AS(1) pointer to `int __attribute__((vector_size(16)))`.
#if __has_builtin(__builtin_amdgcn_global_load_async_to_lds_b128) && \
    __has_builtin(__builtin_amdgcn_s_wait_asynccnt)
#define CDNA5_HAS_ASYNC 1
typedef int v4i_b128 __attribute__((vector_size(16)));
__device__ __forceinline__ void async_b128(const void* g, void* l) {
  __builtin_amdgcn_global_load_async_to_lds_b128(
      (__attribute__((address_space(1))) v4i_b128*)(unsigned long long)g,
      (__attribute__((address_space(3))) v4i_b128*)(unsigned int)(unsigned long long)l,
      0, 0);
}
#endif

__device__ __forceinline__ float gelu_f(float x) {
  const float c0 = 0.7978845608028654f;  // sqrt(2/pi)
  const float c1 = 0.044715f;
  return 0.5f * x * (1.0f + tanhf(c0 * (x + c1 * x * x * x)));
}

// A fragment (16x32 f16, MxK) from row-major f16 [16][ld], tile at column k0.
// Lane L<16 holds row L, K = {k0..k0+7, k0+16..k0+23}; lanes 16..31 get +8.
__device__ __forceinline__ v16h load_a16(const _Float16* m, int ld, int k0) {
  int lane = threadIdx.x & 31;
  int row  = lane & 15;
  int kh   = (lane & 16) ? 8 : 0;
  const _Float16* p = m + row * ld + k0 + kh;
  v8h lo = *(const v8h*)(p);
  v8h hi = *(const v8h*)(p + 16);
  v16h r;
#pragma unroll
  for (int i = 0; i < 8; ++i) { r[i] = lo[i]; r[i + 8] = hi[i]; }
  return r;
}

// B fragment (32x16 f16, KxN) from pre-swizzled weights: per (kt,nt) tile each
// lane owns 16 contiguous halves (32 bytes).
__device__ __forceinline__ v16h load_bswz(const _Float16* w, int ntiles, int kt, int nt) {
  int lane = threadIdx.x & 31;
  const _Float16* p = w + ((kt * ntiles + nt) * 32 + lane) * 16;
  v8h lo = *(const v8h*)(p);
  v8h hi = *(const v8h*)(p + 8);
  v16h r;
#pragma unroll
  for (int i = 0; i < 8; ++i) { r[i] = lo[i]; r[i + 8] = hi[i]; }
  return r;
}

// ---------------------------------------------------------------------------
// f32 [K][N] row-major -> f16 WMMA-B swizzled layout. K%32==0, N%16==0.
// ---------------------------------------------------------------------------
__global__ void convert_swz_kernel(const float* __restrict__ W, _Float16* __restrict__ dst,
                                   int K, int N) {
  int idx = blockIdx.x * 256 + threadIdx.x;
  if (idx >= K * N) return;
  int e    = idx & 15;
  int lane = (idx >> 4) & 31;
  int tile = idx >> 9;
  int ntiles = N >> 4;
  int nt = tile % ntiles;
  int kt = tile / ntiles;
  int col = nt * 16 + (lane & 15);
  int k   = kt * 32 + ((e & 8) ? 16 : 0) + ((lane & 16) ? 8 : 0) + (e & 7);
  dst[idx] = (_Float16)W[k * N + col];
}

// ---------------------------------------------------------------------------
// Generic WMMA GEMM: C[M,N] = act( res + A[M,K] @ Wsw[K,N] + bias )
// 256 threads = 8 waves; block tile 16 rows x 128 cols (one 16-wide N tile per
// wave). REQUIRES M%16==0, K%32==0, N%16==0 (true for every launch below), so
// the staging and epilogue are branch-free.
// ---------------------------------------------------------------------------
__global__ void __launch_bounds__(256)
gemm_kernel(const float* __restrict__ A, const _Float16* __restrict__ Wsw,
            const float* __restrict__ bias, const float* __restrict__ res,
            float* __restrict__ C, int M, int K, int N, int act) {
  __shared__ _Float16 As[16 * 32];
  int tid = threadIdx.x, wave = tid >> 5, lane = tid & 31;
  int row0 = blockIdx.x * 16;
  int n0   = blockIdx.y * 128 + wave * 16;
  int ntiles = N >> 4;
  v8f acc = {};
  // staging coords: each thread owns 2 consecutive halves of the 16x32 tile.
  int sidx = tid * 2;
  int srow = sidx >> 5, skk = sidx & 31;
  for (int k0 = 0; k0 < K; k0 += 32) {
    {
      float2 f = *(const float2*)(A + (size_t)(row0 + srow) * K + k0 + skk);
      _Float16 h0 = (_Float16)f.x, h1 = (_Float16)f.y;
      unsigned u = (unsigned)*(unsigned short*)&h0 |
                   ((unsigned)*(unsigned short*)&h1 << 16);
      *(unsigned*)&As[sidx] = u;
    }
    __syncthreads();
    if (n0 < N) {
      v16h a = load_a16(As, 32, 0);
      v16h b = load_bswz(Wsw, ntiles, k0 >> 5, n0 >> 4);
      acc = WMMA_F16(a, b, acc);
    }
    __syncthreads();
  }
  if (n0 < N) {
    int c  = n0 + (lane & 15);
    int mb = (lane & 16) ? 8 : 0;
#pragma unroll
    for (int i = 0; i < 8; ++i) {
      int r = row0 + mb + i;
      float v = acc[i] + bias[c];
      if (res) v += res[(size_t)r * N + c];
      if (act) v = gelu_f(v);
      C[(size_t)r * N + c] = v;
    }
  }
}

// ---------------------------------------------------------------------------
// Fused equivariant attention: one block = (batch, 16-query tile), 256 threads.
// ---------------------------------------------------------------------------
struct AttnSmem {
  _Float16 w1[128 * 256];   //  64 KB  emb layer-1 weights (swizzled)
  _Float16 w2[256 * 256];   // 128 KB  emb layer-2 weights (swizzled)
  _Float16 rffA[16 * 128];  //   4 KB  RFF features (WMMA A)
  _Float16 hA[16 * 256];    //   8 KB  gelu(l1) activations (WMMA A)
  float emb[16 * 256];      //  16 KB  emb output
  float logits[16][64][4];  //  16 KB  logits -> attention weights
  float qdat[16][4];
  float inv[16][4];
  float relsq[16];
};                          // ~237 KB (< 320 KB per WGP)

__device__ __forceinline__ void emb_gemm(AttnSmem& s, const float* __restrict__ b1,
                                         const float* __restrict__ b2) {
  int tid = threadIdx.x, wave = tid >> 5, lane = tid & 31;
  int n0 = wave * 32;        // 32 output columns per wave (2 N-tiles)
  int c  = lane & 15;
  int mb = (lane & 16) ? 8 : 0;
  v8f zero = {};
  v8f acc0 = zero, acc1 = zero;
#pragma unroll
  for (int kt = 0; kt < 4; ++kt) {   // K = 128
    v16h a  = load_a16(s.rffA, 128, kt * 32);
    v16h bA = load_bswz(s.w1, 16, kt, n0 >> 4);
    v16h bB = load_bswz(s.w1, 16, kt, (n0 >> 4) + 1);
    acc0 = WMMA_F16(a, bA, acc0);
    acc1 = WMMA_F16(a, bB, acc1);
  }
#pragma unroll
  for (int i = 0; i < 8; ++i) {
    float v0 = gelu_f(acc0[i] + b1[n0 + c]);
    float v1 = gelu_f(acc1[i] + b1[n0 + 16 + c]);
    s.hA[(mb + i) * 256 + n0 + c]      = (_Float16)v0;
    s.hA[(mb + i) * 256 + n0 + 16 + c] = (_Float16)v1;
  }
  __syncthreads();
  acc0 = zero; acc1 = zero;
#pragma unroll
  for (int kt = 0; kt < 8; ++kt) {   // K = 256
    v16h a  = load_a16(s.hA, 256, kt * 32);
    v16h bA = load_bswz(s.w2, 16, kt, n0 >> 4);
    v16h bB = load_bswz(s.w2, 16, kt, (n0 >> 4) + 1);
    acc0 = WMMA_F16(a, bA, acc0);
    acc1 = WMMA_F16(a, bB, acc1);
  }
#pragma unroll
  for (int i = 0; i < 8; ++i) {
    s.emb[(mb + i) * 256 + n0 + c]      = acc0[i] + b2[n0 + c];
    s.emb[(mb + i) * 256 + n0 + 16 + c] = acc1[i] + b2[n0 + 16 + c];
  }
  __syncthreads();
}

__global__ void __launch_bounds__(256)
attn_kernel(const float* __restrict__ qsrc, int qstride, const float* __restrict__ p4,
            const float* __restrict__ kbuf, const float* __restrict__ vbuf,
            const float* __restrict__ freqs_q, const _Float16* __restrict__ w1q,
            const float* __restrict__ b1q, const _Float16* __restrict__ w2q,
            const float* __restrict__ b2q,
            const float* __restrict__ freqs_v, const _Float16* __restrict__ w1v,
            const float* __restrict__ b1v, const _Float16* __restrict__ w2v,
            const float* __restrict__ b2v,
            float* __restrict__ outbuf, int Nq, int d_inv, float inv2sig) {
  __shared__ AttnSmem s;
  int tid = threadIdx.x;
  int b = blockIdx.y, q0 = blockIdx.x * 16;

  if (tid < 16) {
    const float* qp = qsrc + ((size_t)b * Nq + q0 + tid) * qstride;
    s.qdat[tid][0] = qp[0];
    s.qdat[tid][1] = qp[1];
    s.qdat[tid][2] = (qstride >= 4) ? qp[2] : 0.f;
    s.qdat[tid][3] = (qstride >= 4) ? qp[3] : 0.f;
  }

  float accq[16];
#pragma unroll
  for (int i = 0; i < 16; ++i) accq[i] = 0.f;

  for (int pass = 0; pass < 2; ++pass) {
    const _Float16* gw1 = pass ? w1v : w1q;
    const _Float16* gw2 = pass ? w2v : w2q;
    const float* gb1 = pass ? b1v : b1q;
    const float* gb2 = pass ? b2v : b2q;
    const float* gfr = pass ? freqs_v : freqs_q;

    // Stage this pass's emb weights into LDS.
#if defined(CDNA5_HAS_ASYNC)
    {
      for (int i = tid; i < (128 * 256) / 8; i += 256)
        async_b128((const char*)gw1 + i * 16, (char*)s.w1 + i * 16);
      for (int i = tid; i < (256 * 256) / 8; i += 256)
        async_b128((const char*)gw2 + i * 16, (char*)s.w2 + i * 16);
      __builtin_amdgcn_s_wait_asynccnt(0);
    }
#else
    {
      const uint4* s1 = (const uint4*)gw1; uint4* d1 = (uint4*)s.w1;
      for (int i = tid; i < (128 * 256) / 8; i += 256) d1[i] = s1[i];
      const uint4* s2 = (const uint4*)gw2; uint4* d2 = (uint4*)s.w2;
      for (int i = tid; i < (256 * 256) / 8; i += 256) d2[i] = s2[i];
    }
#endif
    __syncthreads();

    for (int l = 0; l < 64; ++l) {
      if (tid < 16) {
        const float* lp = p4 + ((size_t)b * 64 + l) * 4;
        float dx = s.qdat[tid][0] - lp[0];
        float dy = s.qdat[tid][1] - lp[1];
        float ck = lp[2], sk = lp[3];
        s.inv[tid][0] = ck * dx + sk * dy;
        s.inv[tid][1] = -sk * dx + ck * dy;
        if (d_inv == 4) {
          float cq = s.qdat[tid][2], sq = s.qdat[tid][3];
          s.inv[tid][2] = cq * ck + sq * sk;
          s.inv[tid][3] = sq * ck - cq * sk;
        } else {
          s.inv[tid][2] = 0.f; s.inv[tid][3] = 0.f;
        }
        s.relsq[tid] = dx * dx + dy * dy;
      }
      __syncthreads();

      // Random Fourier features -> f16 WMMA A tile (16 pairs x 128).
#pragma unroll
      for (int it = 0; it < 4; ++it) {
        int idx = tid + it * 256;
        int pr = idx >> 6, j = idx & 63;
        float pj = 0.f;
        for (int i = 0; i < d_inv; ++i) pj += s.inv[pr][i] * gfr[i * 64 + j];
        pj *= 6.283185307179586f;
        s.rffA[pr * 128 + j]      = (_Float16)cosf(pj);
        s.rffA[pr * 128 + 64 + j] = (_Float16)sinf(pj);
      }
      __syncthreads();

      emb_gemm(s, gb1, gb2);   // -> s.emb[16][256], syncs internally

      if (pass == 0) {
        if (tid < 64) {
          int q = tid >> 2, h = tid & 3;
          const float* kp = kbuf + ((size_t)b * 64 + l) * 256 + h * 64;
          const float* ep = s.emb + q * 256 + h * 64;
          float dot = 0.f;
          for (int d2 = 0; d2 < 64; ++d2) dot += ep[d2] * kp[d2];
          s.logits[q][l][h] = dot * 0.125f - s.relsq[q] * inv2sig;
        }
      } else {
        int c = tid, h = c >> 6;
        if (l + 1 < 64)   // pull next V row toward this WGP's caches
          __builtin_prefetch(vbuf + ((size_t)b * 64 + l + 1) * 256 + c, 0, 0);
        float vv = vbuf[((size_t)b * 64 + l) * 256 + c];
#pragma unroll
        for (int q = 0; q < 16; ++q)
          accq[q] += s.logits[q][l][h] * vv * s.emb[q * 256 + c];
      }
      __syncthreads();
    }

    if (pass == 0) {
      // softmax over latents, per (query, head)
      if (tid < 64) {
        int q = tid >> 2, h = tid & 3;
        float mx = -1e30f;
        for (int l = 0; l < 64; ++l) mx = fmaxf(mx, s.logits[q][l][h]);
        float sum = 0.f;
        for (int l = 0; l < 64; ++l) {
          float e = expf(s.logits[q][l][h] - mx);
          s.logits[q][l][h] = e;
          sum += e;
        }
        float r = 1.f / sum;
        for (int l = 0; l < 64; ++l) s.logits[q][l][h] *= r;
      }
      __syncthreads();
    }
  }

  int c = tid;
#pragma unroll
  for (int q = 0; q < 16; ++q)
    outbuf[((size_t)b * Nq + q0 + q) * 256 + c] = accq[q];
}

// ---------------------------------------------------------------------------
// Small helper kernels.
// ---------------------------------------------------------------------------
__global__ void prep_kernel(const float* __restrict__ p, float* __restrict__ p4, int n) {
  int i = blockIdx.x * blockDim.x + threadIdx.x;
  if (i >= n) return;
  p4[i * 4 + 0] = p[i * 3 + 0];
  p4[i * 4 + 1] = p[i * 3 + 1];
  p4[i * 4 + 2] = cosf(p[i * 3 + 2]);
  p4[i * 4 + 3] = sinf(p[i * 3 + 2]);
}

__global__ void ln_kernel(const float* __restrict__ a, const float* __restrict__ s,
                          const float* __restrict__ b, float* __restrict__ o, int rows) {
  int r = blockIdx.x * blockDim.x + threadIdx.x;
  if (r >= rows) return;
  const float* x = a + r * 64;
  float m = 0.f;
  for (int i = 0; i < 64; ++i) m += x[i];
  m *= (1.f / 64.f);
  float v = 0.f;
  for (int i = 0; i < 64; ++i) { float d = x[i] - m; v += d * d; }
  v *= (1.f / 64.f);
  float inv = rsqrtf(v + 1e-6f);
  for (int i = 0; i < 64; ++i) o[r * 64 + i] = (x[i] - m) * inv * s[i] + b[i];
}

__global__ void out3_kernel(const float* __restrict__ X, const float* __restrict__ W,
                            const float* __restrict__ b, float* __restrict__ out, int rows) {
  int r = blockIdx.x * blockDim.x + threadIdx.x;
  if (r >= rows) return;
  for (int j = 0; j < 3; ++j) {
    float acc = b[j];
    for (int d = 0; d < 64; ++d) acc += X[r * 64 + d] * W[d * 3 + j];
    out[r * 3 + j] = acc;
  }
}

// ---------------------------------------------------------------------------
// Workspace layout
// ---------------------------------------------------------------------------
// f16 weight arena (element offsets)
static const size_t OFF_STEM = 0;                    // 32x64
static const size_t OFF_CK   = 2048;                 // 64x256
static const size_t OFF_CV   = 18432;                // 64x256
static const size_t OFF_CEQ1 = 34816;                // 128x256
static const size_t OFF_CEQ2 = 67584;                // 256x256
static const size_t OFF_CEV1 = 133120;               // 128x256
static const size_t OFF_CEV2 = 165888;               // 256x256
static const size_t OFF_CF1  = 231424;               // 256x256
static const size_t OFF_CF2  = 296960;               // 256x256
static const size_t OFF_O1   = 362496;               // 256x64
static const size_t OFF_O2   = 378880;               // 64x64
static const size_t OFF_SB   = 382976;
static const size_t SB_STRIDE = 253952;
static const size_t HALF_TOTAL = OFF_SB + 2 * SB_STRIDE;  // 890880 halves
// per-self-block relative offsets
static const size_t SBK = 0, SBV = 16384, SBEQ1 = 32768, SBEQ2 = 65536,
                    SBEV1 = 131072, SBEV2 = 163840, SBF1 = 229376, SBF2 = 233472,
                    SBPJ = 237568;
// float arena (element offsets)
static const size_t FP4 = 0, FA = 512, FNORM = 8704, FK = 16896, FV = 49664,
                    FATT = 82432, FH1 = 606720;

extern "C" void kernel_launch(void* const* d_in, const int* in_sizes, int n_in,
                              void* d_out, int out_size, void* d_ws, size_t ws_size,
                              hipStream_t stream) {
  (void)in_sizes; (void)n_in; (void)out_size; (void)ws_size;

  _Float16* hw = (_Float16*)d_ws;
  float* wf = (float*)((char*)d_ws + HALF_TOTAL * sizeof(_Float16));
  float* p4  = wf + FP4;
  float* aB  = wf + FA;
  float* an  = wf + FNORM;
  float* kB  = wf + FK;
  float* vB  = wf + FV;
  float* atO = wf + FATT;
  float* h1  = wf + FH1;

  auto FIN = [&](int i) { return (const float*)d_in[i]; };
  auto conv = [&](int idx, int K, int N, size_t off) {
    int total = K * N;
    convert_swz_kernel<<<dim3((total + 255) / 256), dim3(256), 0, stream>>>(
        FIN(idx), hw + off, K, N);
  };
  auto gemm = [&](const float* A, size_t woff, int bidx, const float* res, float* C,
                  int M, int K, int N, int act) {
    dim3 g((unsigned)((M + 15) / 16), (unsigned)((N + 127) / 128));
    gemm_kernel<<<g, dim3(256), 0, stream>>>(A, hw + woff, FIN(bidx), res, C, M, K, N, act);
  };

  // ---- convert all WMMA weights to swizzled f16 ----
  conv(73, 32, 64, OFF_STEM);
  conv(17, 64, 256, OFF_CK);   conv(21, 64, 256, OFF_CV);
  conv(4, 128, 256, OFF_CEQ1); conv(6, 256, 256, OFF_CEQ2);
  conv(9, 128, 256, OFF_CEV1); conv(11, 256, 256, OFF_CEV2);
  conv(13, 256, 256, OFF_CF1); conv(15, 256, 256, OFF_CF2);
  conv(23, 256, 64, OFF_O1);   conv(25, 64, 64, OFF_O2);
  for (int i = 0; i < 2; ++i) {
    int base = 29 + 22 * i;
    size_t sb = OFF_SB + SB_STRIDE * (size_t)i;
    conv(base + 14, 64, 256, sb + SBK);
    conv(base + 20, 64, 256, sb + SBV);
    conv(base + 1, 128, 256, sb + SBEQ1);
    conv(base + 3, 256, 256, sb + SBEQ2);
    conv(base + 6, 128, 256, sb + SBEV1);
    conv(base + 8, 256, 256, sb + SBEV2);
    conv(base + 10, 64, 64, sb + SBF1);
    conv(base + 12, 64, 64, sb + SBF2);
    conv(base + 18, 256, 64, sb + SBPJ);
  }

  // ---- p -> (x, y, cos, sin); stem ----
  prep_kernel<<<dim3(1), dim3(128), 0, stream>>>(FIN(1), p4, 128);
  gemm(FIN(2), OFF_STEM, 74, nullptr, aB, 128, 32, 64, 0);

  // ---- self-attention blocks ----
  for (int i = 0; i < 2; ++i) {
    int base = 29 + 22 * i;
    size_t sb = OFF_SB + SB_STRIDE * (size_t)i;
    ln_kernel<<<dim3(1), dim3(128), 0, stream>>>(aB, FIN(base + 16), FIN(base + 17), an, 128);
    gemm(an, sb + SBK, base + 15, nullptr, kB, 128, 64, 256, 0);
    gemm(an, sb + SBV, base + 21, nullptr, vB, 128, 64, 256, 0);
    attn_kernel<<<dim3(4, 2), dim3(256), 0, stream>>>(
        p4, 4, p4, kB, vB,
        FIN(base + 0), hw + sb + SBEQ1, FIN(base + 2), hw + sb + SBEQ2, FIN(base + 4),
        FIN(base + 5), hw + sb + SBEV1, FIN(base + 7), hw + sb + SBEV2, FIN(base + 9),
        atO, 64, 4, 0.5f);
    gemm(atO, sb + SBPJ, base + 19, aB, an, 128, 256, 64, 0);    // h = a + proj(attn)
    gemm(an, sb + SBF1, base + 11, nullptr, kB, 128, 64, 64, 1); // t = gelu(h@ffn1+b)
    gemm(kB, sb + SBF2, base + 13, aB, aB, 128, 64, 64, 1);      // a = gelu(a + t@ffn2+b)
  }

  // ---- cross-attention block + head ----
  ln_kernel<<<dim3(1), dim3(128), 0, stream>>>(aB, FIN(19), FIN(20), an, 128);
  gemm(an, OFF_CK, 18, nullptr, kB, 128, 64, 256, 0);
  gemm(an, OFF_CV, 22, nullptr, vB, 128, 64, 256, 0);
  attn_kernel<<<dim3(64, 2), dim3(256), 0, stream>>>(
      FIN(0), 2, p4, kB, vB,
      FIN(3), hw + OFF_CEQ1, FIN(5), hw + OFF_CEQ2, FIN(7),
      FIN(8), hw + OFF_CEV1, FIN(10), hw + OFF_CEV2, FIN(12),
      atO, 1024, 2, 0.5f);
  gemm(atO, OFF_CF1, 14, nullptr, h1, 2048, 256, 256, 1);   // gelu(ffn1)
  gemm(h1, OFF_CF2, 16, nullptr, atO, 2048, 256, 256, 1);   // gelu(ffn2(.))
  gemm(atO, OFF_O1, 24, nullptr, h1, 2048, 256, 64, 1);     // gelu(out1)
  gemm(h1, OFF_O2, 26, nullptr, atO, 2048, 64, 64, 1);      // gelu(out2)
  out3_kernel<<<dim3(16), dim3(128), 0, stream>>>(atO, FIN(27), FIN(28),
                                                  (float*)d_out, 2048);
}
